// CIGAR_WO_CDGNN_89026082111523
// MI455X (gfx1250) — matrix operand order repaired and checked by reading, not compiled, not run on hardware
//
#include <hip/hip_runtime.h>
#include <hip/hip_bf16.h>
#include <math.h>

typedef __attribute__((ext_vector_type(16))) _Float16     v16h;
typedef __attribute__((ext_vector_type(8)))  float        v8f;
typedef __attribute__((ext_vector_type(4)))  unsigned int v4u;
typedef __attribute__((ext_vector_type(8)))  int          v8i;
typedef __attribute__((ext_vector_type(4)))  int          v4i;

#define BATCH 512
#define SEQL  200
#define KDIM  64
#define PTD   256   // prototype / pt dim
#define NPROT 16
#define PANEL 128   // N-columns per block in the WMMA precompute
#define LDSROW 129  // padded LDS row stride (floats) -> bank-conflict-free frag reads

// ---------------------------------------------------------------------------
// Kernel 1: embedding gathers, masked mean pooling, build x = [user|item|mean|item*mean]
// one block per sample, 256 threads (thread t -> table t>>6, col t&63)
// ---------------------------------------------------------------------------
__global__ void embed_pool_kernel(
    const int* __restrict__ uf0, const int* __restrict__ uf1, const int* __restrict__ uf2,
    const int* __restrict__ if0, const int* __restrict__ if1, const int* __restrict__ if2, const int* __restrict__ if3,
    const int* __restrict__ s0,  const int* __restrict__ s1,  const int* __restrict__ s2,  const int* __restrict__ s3,
    const float* __restrict__ ut0, const float* __restrict__ ut1, const float* __restrict__ ut2,
    const float* __restrict__ it0, const float* __restrict__ it1, const float* __restrict__ it2, const float* __restrict__ it3,
    float* __restrict__ x /* [512,960] */)
{
    const int b   = blockIdx.x;
    const int t   = threadIdx.x;       // 0..255
    const int tab = t >> 6;
    const int col = t & 63;

    // user embedding -> x[:, 0:192]
    if (t < 192) {
        const float* utab = (tab == 0) ? ut0 : (tab == 1 ? ut1 : ut2);
        const int*   uf   = (tab == 0) ? uf0 : (tab == 1 ? uf1 : uf2);
        x[(size_t)b * 960 + t] = utab[(size_t)uf[b] * KDIM + col];
    }

    const float* itab = (tab == 0) ? it0 : (tab == 1 ? it1 : (tab == 2 ? it2 : it3));
    const int*   ifv  = (tab == 0) ? if0 : (tab == 1 ? if1 : (tab == 2 ? if2 : if3));
    const int*   sv   = (tab == 0) ? s0  : (tab == 1 ? s1  : (tab == 2 ? s2  : s3));

    const float item = itab[(size_t)ifv[b] * KDIM + col];

    const int* m0 = s0 + (size_t)b * SEQL;   // mask source (item_f0_seq)
    const int* si = sv + (size_t)b * SEQL;
    float acc = 0.0f;
    int   cnt = 0;
    for (int l = 0; l < SEQL; ++l) {
        if (l + 1 < SEQL)   // hide gather latency: prefetch next row of this table
            __builtin_prefetch(itab + (size_t)si[l + 1] * KDIM + col, 0, 3);
        const int mval = m0[l];              // uniform across the block
        if (mval != 0) {
            acc += itab[(size_t)si[l] * KDIM + col];
            ++cnt;
        }
    }
    const float mean = acc / fmaxf((float)cnt, 1.0f);

    x[(size_t)b * 960 + 192 + t] = item;
    x[(size_t)b * 960 + 448 + t] = mean;
    x[(size_t)b * 960 + 704 + t] = item * mean;
}

// ---------------------------------------------------------------------------
// Kernel 2a: L2-normalize the 16 prototype rows. block r -> row r.
// ---------------------------------------------------------------------------
__global__ void proto_norm_kernel(const float* __restrict__ protos,
                                  float* __restrict__ protos_n)
{
    __shared__ float red[256];
    const int r = blockIdx.x, t = threadIdx.x;
    const float v = protos[r * PTD + t];
    red[t] = v * v;
    __syncthreads();
    for (int s = 128; s > 0; s >>= 1) {
        if (t < s) red[t] += red[t + s];
        __syncthreads();
    }
    protos_n[r * PTD + t] = v * rsqrtf(red[0]);
}

// ---------------------------------------------------------------------------
// Kernel 2b: per-sample nearest prototype by cosine sim.
// argmax(item_emb_hat . protos_n) == argmax(item_emb . protos_n)  (positive scale)
// ---------------------------------------------------------------------------
__global__ void proto_assign_kernel(const float* __restrict__ x,
                                    const float* __restrict__ protos_n,
                                    int* __restrict__ gidx)
{
    __shared__ float red[256];
    __shared__ float sims[NPROT];
    const int b = blockIdx.x, t = threadIdx.x;
    const float e = x[(size_t)b * 960 + 192 + t];    // item_emb column t
    for (int pr = 0; pr < NPROT; ++pr) {
        red[t] = e * protos_n[pr * PTD + t];
        __syncthreads();
        for (int s = 128; s > 0; s >>= 1) {
            if (t < s) red[t] += red[t + s];
            __syncthreads();
        }
        if (t == 0) sims[pr] = red[0];
        __syncthreads();
    }
    if (t == 0) {
        int best = 0; float bv = sims[0];
        for (int pr = 1; pr < NPROT; ++pr)
            if (sims[pr] > bv) { bv = sims[pr]; best = pr; }   // first-max tie break
        gidx[b] = best;
    }
}

// ---------------------------------------------------------------------------
// Kernel 3 (WMMA + TDM): Out[16,N] = tanh(protos_n[16,256] @ W[256,N] + bias[N]) * scale[N]
// REQUIRES N % 128 == 0. One 128-col panel per 256-thread block (8 waves, 1 tile each).
// K = 256 -> 8 chunks of 32. Each [32 x 128] f32 W-chunk is DMA'd into LDS by the
// Tensor Data Mover (double-buffered, padded to a 129-float row stride), overlapped
// with WMMA on the previous chunk via TENSORcnt + block barriers.
// ---------------------------------------------------------------------------
__global__ void hyper_precompute_kernel(const float* __restrict__ P,      // protos_n [16,256]
                                        const float* __restrict__ W,      // [256, N]
                                        const float* __restrict__ bias,   // [N]
                                        const float* __restrict__ scale,  // [N]
                                        float* __restrict__ Out,          // [16, N]
                                        int N)
{
    __shared__ float lds_buf[2][32 * LDSROW];     // 2 x 16.5 KB

    const int  lane   = threadIdx.x & 31;
    const int  wv     = threadIdx.x >> 5;         // 0..7 : tile within panel
    const int  m      = lane & 15;
    const int  kbA    = (lane >> 4) * 8;          // A: lanes16-31 -> K +8 within half
    const int  kbB    = (lane >> 4) * 16;         // unused: B fragments below index with kbA
    const int  nl     = wv * 16 + (lane & 15);    // column within panel (0..127)
    const int  n      = blockIdx.x * PANEL + nl;  // global output column
    const bool leader = (threadIdx.x < 32);       // wave 0 drives the TDM

    // ---- A fragments (16x32 f16), hoisted: protos_n reused by every tile ----
    v16h afrag[8];
    #pragma unroll
    for (int kk = 0; kk < 8; ++kk) {
        const float* Pa = P + m * PTD + kk * 32;
        #pragma unroll
        for (int j = 0; j < 16; ++j) {
            const int v  = j >> 1, pos = j & 1;
            const int kl = (v < 4) ? (2 * v + pos) : (16 + 2 * (v - 4) + pos);
            afrag[kk][j] = (_Float16)Pa[kl + kbA];
        }
    }

    // ---- TDM descriptor pieces (all wave-uniform scalars) ----
    const unsigned long long wbase =
        (unsigned long long)(uintptr_t)W + (unsigned long long)blockIdx.x * (PANEL * 4u);
    const unsigned long long kstride = (unsigned long long)N * 128u;   // 32 rows * N * 4B

    auto tdm_issue = [&](int kk, int buf) {
        const unsigned long long ga = wbase + (unsigned long long)kk * kstride;
        const unsigned int ldsa = (unsigned int)(uintptr_t)(void*)&lds_buf[buf][0];
        v4u g0; v8i g1; v4i gz; v8i gz8;
        g0[0] = 1u;                                   // count=1, no gather
        g0[1] = ldsa;                                 // lds_addr (bytes)
        g0[2] = (unsigned int)ga;                     // global_addr[31:0]
        g0[3] = (unsigned int)(ga >> 32) | (2u << 30);// global_addr[56:32] | type=2
        // data_size=4B(2), pad_enable, pad_interval=128 DW(6), pad_amount=1 DW(0)
        g1[0] = (int)((2u << 16) | (1u << 20) | (6u << 22));
        g1[1] = (int)(((unsigned)N & 0xFFFFu) << 16);               // tensor_dim0[15:0]
        g1[2] = (int)((((unsigned)N >> 16) & 0xFFFFu) | (256u << 16)); // dim0 hi | tensor_dim1=256
        g1[3] = (int)(128u << 16);                                  // tile_dim0=128
        g1[4] = (int)32u;                                           // tile_dim1=32, tile_dim2=0
        g1[5] = (int)(unsigned)N;                                   // tensor_dim0_stride lo32
        g1[6] = 0;                                                  // stride hi | dim1_stride lo
        g1[7] = 0;
        gz[0] = gz[1] = gz[2] = gz[3] = 0;                          // groups 2/3 unused (2-D tile)
        gz8[0] = gz8[1] = gz8[2] = gz8[3] = 0;
        gz8[4] = gz8[5] = gz8[6] = gz8[7] = 0;
        __builtin_amdgcn_tensor_load_to_lds(g0, g1, gz, gz, gz8, 0);
    };

    if (leader) tdm_issue(0, 0);

    v8f acc = {};
    #pragma unroll
    for (int kk = 0; kk < 8; ++kk) {
        if (leader) __builtin_amdgcn_s_wait_tensorcnt(0);   // chunk kk resident
        __syncthreads();                                    // publish LDS to all waves
        if (kk < 7 && leader) tdm_issue(kk + 1, (kk + 1) & 1);  // overlap next transfer

        const float* Bt = &lds_buf[kk & 1][0];
        v16h bfrag;
        #pragma unroll
        for (int j = 0; j < 16; ++j)                        // conflict-free: stride 129
            bfrag[j] = (_Float16)Bt[(kbA + (j < 8 ? j : j + 8)) * LDSROW + nl];

        acc = __builtin_amdgcn_wmma_f32_16x16x32_f16(
                  false, afrag[kk], false, bfrag, (short)0, acc, false, false);
    }

    // ---- epilogue: D layout -> lane holds col n, rows r + 8*(lane>=16) ----
    const float bv  = bias[n];
    const float sv  = scale[n];
    const int   mhi = (lane >> 4) * 8;
    #pragma unroll
    for (int r = 0; r < 8; ++r)
        Out[(size_t)(r + mhi) * N + n] = tanhf(acc[r] + bv) * sv;
}

// ---------------------------------------------------------------------------
// Kernel 3b: ragged fallback for tiny N (layer-2 bias, N==1).
// block g -> prototype g; 256-thread dot + reduction per column.
// ---------------------------------------------------------------------------
__global__ void hyper_small_kernel(const float* __restrict__ P,
                                   const float* __restrict__ W,
                                   const float* __restrict__ bias,
                                   const float* __restrict__ scale,
                                   float* __restrict__ Out, int N)
{
    __shared__ float red[256];
    const int g = blockIdx.x, t = threadIdx.x;
    const float pv = P[g * PTD + t];
    for (int nn = 0; nn < N; ++nn) {
        red[t] = pv * W[(size_t)t * N + nn];
        __syncthreads();
        for (int s = 128; s > 0; s >>= 1) {
            if (t < s) red[t] += red[t + s];
            __syncthreads();
        }
        if (t == 0) Out[(size_t)g * N + nn] = tanhf(red[0] + bias[nn]) * scale[nn];
        __syncthreads();
    }
}

// ---------------------------------------------------------------------------
// Kernel 4: per-sample layer apply.
//   origin' = lW @ origin + lb            (shared weights)
//   group'  = Wt[g] @ group + bvec[g]     (Wt pre-scaled by lW, L2-resident)
// One block per sample; one wave per output row; lanes stride the p dim.
// ---------------------------------------------------------------------------
__global__ void apply_layer_kernel(const float* __restrict__ origin_in, // [B,p]
                                   const float* __restrict__ group_in,  // [B,p]
                                   const float* __restrict__ lW,        // [h,p]
                                   const float* __restrict__ lb,        // [h]
                                   const float* __restrict__ Wt,        // [16,h,p]
                                   const float* __restrict__ bvec,      // [16,h]
                                   const int*   __restrict__ gidx,      // [B]
                                   float* __restrict__ origin_out,
                                   float* __restrict__ group_out,
                                   int p, int h, int relu, int ostride)
{
    const int b    = blockIdx.x;
    const int lane = threadIdx.x & 31;
    const int wid  = threadIdx.x >> 5;
    const int nw   = blockDim.x >> 5;
    const int g    = gidx[b];

    const float* og = origin_in + (size_t)b * p;
    const float* gr = group_in  + (size_t)b * p;
    const float* Wg = Wt + (size_t)g * h * p;

    for (int hh = wid; hh < h; hh += nw) {
        const float* wo = lW + (size_t)hh * p;
        const float* wg = Wg + (size_t)hh * p;
        float so = 0.0f, sg = 0.0f;
        for (int a = lane; a < p; a += 32) {
            so += wo[a] * og[a];
            sg += wg[a] * gr[a];
        }
        #pragma unroll
        for (int off = 16; off > 0; off >>= 1) {
            so += __shfl_xor(so, off, 32);
            sg += __shfl_xor(sg, off, 32);
        }
        if (lane == 0) {
            float ro = so + lb[hh];
            float rg = sg + bvec[(size_t)g * h + hh];
            if (relu) { ro = fmaxf(ro, 0.0f); rg = fmaxf(rg, 0.0f); }
            origin_out[(size_t)b * ostride + hh] = ro;
            group_out [(size_t)b * ostride + hh] = rg;
        }
    }
}

// ---------------------------------------------------------------------------
// workspace layout (float offsets, all 16-aligned)
// ---------------------------------------------------------------------------
#define WS_PROTOS_N 0u                          // 16*256
#define WS_X        4096u                       // 512*960
#define WS_WT0      495616u                     // 16*256*960
#define WS_B0       4427776u                    // 16*256
#define WS_WT1      4431872u                    // 16*128*256
#define WS_B1       4956160u                    // 16*128
#define WS_WT2      4958208u                    // 16*1*128
#define WS_B2       4960256u                    // 16 (+pad)
#define WS_O0       4960272u                    // 512*256
#define WS_G0       5091344u                    // 512*256
#define WS_O1       5222416u                    // 512*128
#define WS_G1       5287952u                    // 512*128
#define WS_GIDX     5353488u                    // 512 ints

extern "C" void kernel_launch(void* const* d_in, const int* in_sizes, int n_in,
                              void* d_out, int out_size, void* d_ws, size_t ws_size,
                              hipStream_t stream) {
    const int*   uf0 = (const int*)d_in[0];
    const int*   uf1 = (const int*)d_in[1];
    const int*   uf2 = (const int*)d_in[2];
    const int*   if0 = (const int*)d_in[3];
    const int*   if1 = (const int*)d_in[4];
    const int*   if2 = (const int*)d_in[5];
    const int*   if3 = (const int*)d_in[6];
    const int*   s0  = (const int*)d_in[7];
    const int*   s1  = (const int*)d_in[8];
    const int*   s2  = (const int*)d_in[9];
    const int*   s3  = (const int*)d_in[10];
    const float* ut0 = (const float*)d_in[11];
    const float* ut1 = (const float*)d_in[12];
    const float* ut2 = (const float*)d_in[13];
    const float* it0 = (const float*)d_in[14];
    const float* it1 = (const float*)d_in[15];
    const float* it2 = (const float*)d_in[16];
    const float* it3 = (const float*)d_in[17];
    const float* protos = (const float*)d_in[18];
    // per-layer: wtb_W, wtb_b, wtw_W, wtw_b, lin_W, lin_b
    const float* bW[3]; const float* bb[3]; const float* wW[3];
    const float* wb[3]; const float* lW[3]; const float* lb[3];
    for (int i = 0; i < 3; ++i) {
        bW[i] = (const float*)d_in[19 + 6 * i + 0];
        bb[i] = (const float*)d_in[19 + 6 * i + 1];
        wW[i] = (const float*)d_in[19 + 6 * i + 2];
        wb[i] = (const float*)d_in[19 + 6 * i + 3];
        lW[i] = (const float*)d_in[19 + 6 * i + 4];
        lb[i] = (const float*)d_in[19 + 6 * i + 5];
    }
    float* out = (float*)d_out;
    float* ws  = (float*)d_ws;

    float* protos_n = ws + WS_PROTOS_N;
    float* X        = ws + WS_X;
    float* WT[3]    = { ws + WS_WT0, ws + WS_WT1, ws + WS_WT2 };
    float* BV[3]    = { ws + WS_B0,  ws + WS_B1,  ws + WS_B2  };
    float* O0 = ws + WS_O0; float* G0 = ws + WS_G0;
    float* O1 = ws + WS_O1; float* G1 = ws + WS_G1;
    int*   gidx = (int*)(ws + WS_GIDX);

    // 1) embeddings + pooling + x
    embed_pool_kernel<<<BATCH, 256, 0, stream>>>(
        uf0, uf1, uf2, if0, if1, if2, if3, s0, s1, s2, s3,
        ut0, ut1, ut2, it0, it1, it2, it3, X);

    // 2) prototype normalize + assignment
    proto_norm_kernel<<<NPROT, 256, 0, stream>>>(protos, protos_n);
    proto_assign_kernel<<<BATCH, 256, 0, stream>>>(X, protos_n, gidx);

    // 3) WMMA+TDM hypernet precompute per prototype (w-matrix + b-vector per layer)
    const int HID[3]  = { 256, 128, 1 };
    const int PREV[3] = { 960, 256, 128 };
    for (int i = 0; i < 3; ++i) {
        const int Nw = PREV[i] * HID[i];            // 245760 / 32768 / 128 : all %128==0
        hyper_precompute_kernel<<<Nw / PANEL, 256, 0, stream>>>(
            protos_n, wW[i], wb[i], lW[i], WT[i], Nw);
        const int Nb = HID[i];                      // 256 / 128 / 1
        if (Nb % PANEL == 0) {
            hyper_precompute_kernel<<<Nb / PANEL, 256, 0, stream>>>(
                protos_n, bW[i], bb[i], lb[i], BV[i], Nb);
        } else {
            hyper_small_kernel<<<NPROT, 256, 0, stream>>>(
                protos_n, bW[i], bb[i], lb[i], BV[i], Nb);
        }
    }

    // 4) per-sample MLP apply (origin + group), final layer interleaves into d_out[B,2]
    apply_layer_kernel<<<BATCH, 256, 0, stream>>>(
        X, X, lW[0], lb[0], WT[0], BV[0], gidx, O0, G0, 960, 256, 1, 256);
    apply_layer_kernel<<<BATCH, 256, 0, stream>>>(
        O0, G0, lW[1], lb[1], WT[1], BV[1], gidx, O1, G1, 256, 128, 1, 128);
    apply_layer_kernel<<<BATCH, 32, 0, stream>>>(
        O1, G1, lW[2], lb[2], WT[2], BV[2], gidx, out, out + 1, 128, 1, 0, 2);

    (void)in_sizes; (void)n_in; (void)out_size; (void)ws_size;
}
